// SS3D_Seq_62483184222323
// MI455X (gfx1250) — compile-verified, matrix operand order checked by reading, not run on hardware
//
#include <hip/hip_runtime.h>
#include <hip/hip_bf16.h>

typedef __attribute__((ext_vector_type(16))) _Float16 v16h;
typedef __attribute__((ext_vector_type(8)))  _Float16 v8h;
typedef __attribute__((ext_vector_type(8)))  float    v8f;

#define L_TOT   13824   // 24^3
#define SS      24
#define DI      128
#define DSTATE  16
#define DTR     8
#define NCHUNK  864     // L/16
#define XROWS   40      // DT_RANK + 2*D_STATE

// ---------- direction permutation: sequence index l' -> spatial index ----------
__device__ __forceinline__ int perm_l(int dir, int lp) {
    int lf = (dir & 1) ? (L_TOT - 1 - lp) : lp;
    int a0 = lf / 576;
    int a1 = (lf / 24) % 24;
    int a2 = lf % 24;
    int h, w, dd;
    switch (dir >> 1) {
        case 0:  w = a0; h = a1; dd = a2; break;   // swap(2,3): (W,H,D)
        case 1:  h = a0; w = a1; dd = a2; break;   // identity : (H,W,D)
        case 2:  dd = a0; w = a1; h = a2; break;   // swap(2,4): (D,W,H)
        default: h = a0; dd = a1; w = a2; break;   // swap(3,4): (H,D,W)
    }
    return (h * 24 + w) * 24 + dd;
}

__device__ __forceinline__ float softplus_f(float x) {
    return (x > 20.f) ? x : log1pf(__expf(x));
}
__device__ __forceinline__ float silu_f(float x) {
    return x / (1.f + __expf(-x));
}

// Pack two contiguous 8-float runs (k = base..base+7 and base+16..base+23 per
// CDNA5 16-bit A/B operand layout) into a v16h WMMA operand.
__device__ __forceinline__ v16h pack_f32(const float4* row4, int kk, int g) {
    float4 a0 = row4[kk * 8 + 2 * g];
    float4 a1 = row4[kk * 8 + 2 * g + 1];
    float4 a2 = row4[kk * 8 + 4 + 2 * g];
    float4 a3 = row4[kk * 8 + 4 + 2 * g + 1];
    v16h r;
    r[0] = (_Float16)a0.x; r[1] = (_Float16)a0.y; r[2] = (_Float16)a0.z; r[3] = (_Float16)a0.w;
    r[4] = (_Float16)a1.x; r[5] = (_Float16)a1.y; r[6] = (_Float16)a1.z; r[7] = (_Float16)a1.w;
    r[8] = (_Float16)a2.x; r[9] = (_Float16)a2.y; r[10] = (_Float16)a2.z; r[11] = (_Float16)a2.w;
    r[12] = (_Float16)a3.x; r[13] = (_Float16)a3.y; r[14] = (_Float16)a3.z; r[15] = (_Float16)a3.w;
    return r;
}

// ---------------- in_proj: xz[l,e] = sum_c x[l,c]*W[e,c]; split xp / z --------
__global__ void k_inproj(const float* __restrict__ x, const float* __restrict__ W,
                         float* __restrict__ xp, float* __restrict__ zb) {
    int tile  = blockIdx.x;                 // 864 M-tiles of 16
    int wave  = threadIdx.x >> 5;
    int lane  = threadIdx.x & 31;
    int ntile = blockIdx.y * 4 + wave;      // 0..15 (N=256)
    int m = lane & 15, g = lane >> 4;
    const float4* Arow = (const float4*)(x + (tile * 16 + m) * 128);
    const float4* Brow = (const float4*)(W + (ntile * 16 + m) * 128);
    v8f acc = {};
#pragma unroll
    for (int kk = 0; kk < 4; ++kk) {
        v16h a = pack_f32(Arow, kk, g);
        v16h b = pack_f32(Brow, kk, g);
        acc = __builtin_amdgcn_wmma_f32_16x16x32_f16(false, a, false, b,
                                                     (short)0, acc, false, false);
    }
    // wave-uniform split: ntile<8 -> xp, else -> z
    int lbase = tile * 16 + 8 * g;
    if (ntile < 8) {
        float* dst = xp + lbase * 128 + ntile * 16 + m;
#pragma unroll
        for (int r = 0; r < 8; ++r) dst[r * 128] = acc[r];
    } else {
        float* dst = zb + lbase * 128 + (ntile - 8) * 16 + m;
#pragma unroll
        for (int r = 0; r < 8; ++r) dst[r * 128] = acc[r];
    }
}

// ---------------- depthwise 3x3x3 conv + bias + SiLU -> xc[l][c] -------------
__global__ void k_conv(const float* __restrict__ xp, const float* __restrict__ cw,
                       const float* __restrict__ cb, float* __restrict__ xc) {
    int l = blockIdx.x, c = threadIdx.x;
    int dd = l % 24, w_ = (l / 24) % 24, h = l / 576;
    float acc = cb[c];
#pragma unroll
    for (int kz = 0; kz < 3; ++kz)
#pragma unroll
        for (int ky = 0; ky < 3; ++ky)
#pragma unroll
            for (int kx = 0; kx < 3; ++kx) {
                int hh = h + kz - 1, ww = w_ + ky - 1, d2 = dd + kx - 1;
                if (hh >= 0 && hh < SS && ww >= 0 && ww < SS && d2 >= 0 && d2 < SS)
                    acc += cw[c * 27 + (kz * 3 + ky) * 3 + kx] *
                           xp[((hh * 24 + ww) * 24 + d2) * 128 + c];
            }
    xc[l * 128 + c] = silu_f(acc);
}

// ---- per-direction: gather xs, x_dbl GEMM (WMMA), dt proj, extract B/C ------
__global__ void k_proj(int dir, const float* __restrict__ xc, float* __restrict__ xs,
                       const float* __restrict__ xpw, const float* __restrict__ dtw,
                       float* __restrict__ dtv, float* __restrict__ Bb,
                       float* __restrict__ Cb) {
    __shared__ _Float16 Ah[16][128];
    __shared__ float    xdbl[16][48];
    __shared__ int      sperm[16];
    int t = threadIdx.x, tile = blockIdx.x;
    if (t < 16) sperm[t] = perm_l(dir, tile * 16 + t);
    __syncthreads();
    // gather one 16x128 activation tile (and persist sequence-ordered xs)
#pragma unroll 4
    for (int r = 0; r < 16; ++r) {
        float v = xc[sperm[r] * 128 + t];
        xs[(tile * 16 + r) * 128 + t] = v;
        Ah[r][t] = (_Float16)v;
    }
    __syncthreads();
    int wave = t >> 5, lane = t & 31;
    if (wave < 3) {                         // 3 N-tiles cover 40 rows (padded to 48)
        int m = lane & 15, g = lane >> 4;
        int cg = wave * 16 + m;
        const float4* Brow = (const float4*)(xpw + cg * 128);
        v8f acc = {};
#pragma unroll
        for (int kk = 0; kk < 4; ++kk) {
            // A operand: two 16B vector loads from LDS (contiguous K runs)
            v8h lo = *(const v8h*)&Ah[m][kk * 32 + 8 * g];
            v8h hi = *(const v8h*)&Ah[m][kk * 32 + 16 + 8 * g];
            v16h a = __builtin_shufflevector(lo, hi, 0, 1, 2, 3, 4, 5, 6, 7,
                                             8, 9, 10, 11, 12, 13, 14, 15);
            v16h b;
            if (cg < XROWS) b = pack_f32(Brow, kk, g);
            else { v16h zz = {}; b = zz; }
            acc = __builtin_amdgcn_wmma_f32_16x16x32_f16(false, a, false, b,
                                                         (short)0, acc, false, false);
        }
#pragma unroll
        for (int r = 0; r < 8; ++r)
            xdbl[r + 8 * g][wave * 16 + m] = acc[r];
    }
    __syncthreads();
    // dt projection: dtv[l,d] = sum_r xdbl[l,r]*dtw[d,r]
    {
        int d = t;
        float wreg[DTR];
#pragma unroll
        for (int r = 0; r < DTR; ++r) wreg[r] = dtw[d * DTR + r];
        for (int row = 0; row < 16; ++row) {
            float s = 0.f;
#pragma unroll
            for (int r = 0; r < DTR; ++r) s += xdbl[row][r] * wreg[r];
            dtv[(tile * 16 + row) * 128 + d] = s;
        }
    }
    // B/C extraction: 256 values each, 2 per thread
#pragma unroll
    for (int s = 0; s < 2; ++s) {
        int id = t * 2 + s;
        int row = id >> 4, n = id & 15;
        Bb[(tile * 16 + row) * DSTATE + n] = xdbl[row][DTR + n];
        Cb[(tile * 16 + row) * DSTATE + n] = xdbl[row][DTR + DSTATE + n];
    }
}

// ---- scan pass 1: per-chunk aggregates (Ac, Bc) for each (d,n) --------------
__global__ void k_scan1(int dir, const float* __restrict__ xs, const float* __restrict__ dtv,
                        const float* __restrict__ Bb, const float* __restrict__ A_logs,
                        const float* __restrict__ dtb, float* __restrict__ aggA,
                        float* __restrict__ aggB) {
    __shared__ float Bs[256];
    int d = threadIdx.x, chunk = blockIdx.x, base = chunk * 16;
    Bs[d]       = Bb[base * DSTATE + d];
    Bs[128 + d] = Bb[base * DSTATE + 128 + d];
    float A[DSTATE];
#pragma unroll
    for (int n = 0; n < DSTATE; ++n) A[n] = -__expf(A_logs[(dir * 128 + d) * DSTATE + n]);
    float bias = dtb[dir * 128 + d];
    float aP[DSTATE], bA[DSTATE];
#pragma unroll
    for (int n = 0; n < DSTATE; ++n) { aP[n] = 1.f; bA[n] = 0.f; }
    __syncthreads();
    for (int t = 0; t < 16; ++t) {
        int l = base + t;
        float dt = softplus_f(dtv[l * 128 + d] + bias);
        float du = dt * xs[l * 128 + d];
#pragma unroll
        for (int n = 0; n < DSTATE; ++n) {
            float a = __expf(dt * A[n]);
            aP[n] = a * aP[n];
            bA[n] = a * bA[n] + du * Bs[t * 16 + n];
        }
    }
#pragma unroll
    for (int n = 0; n < DSTATE; ++n) {
        int idx = (d * DSTATE + n) * NCHUNK + chunk;
        aggA[idx] = aP[n];
        aggB[idx] = bA[n];
    }
}

// ---- scan pass 2: serial scan over 864 chunk aggregates, 2048-way parallel --
__global__ void k_chunkscan(const float* __restrict__ aggA, float* __restrict__ aggB) {
    int dn = blockIdx.x * 256 + threadIdx.x;   // 0..2047
    int base = dn * NCHUNK;
    float h = 0.f;
    for (int c = 0; c < NCHUNK; ++c) {
        // keep the serial pointer-chase fed from L2 (global_prefetch_b8)
        if ((c & 31) == 0 && c + 64 < NCHUNK) {
            __builtin_prefetch(&aggA[base + c + 64], 0, 1);
            __builtin_prefetch(&aggB[base + c + 64], 1, 1);
        }
        float a = aggA[base + c], b = aggB[base + c];
        aggB[base + c] = h;                    // store carry-IN for pass 3
        h = a * h + b;
    }
}

// ---- scan pass 3: replay chunk from carry-in, emit y, scatter into xc -------
__global__ void k_scan3(int dir, const float* __restrict__ xs, const float* __restrict__ dtv,
                        const float* __restrict__ Bb, const float* __restrict__ Cb,
                        const float* __restrict__ A_logs, const float* __restrict__ dtb,
                        const float* __restrict__ Dsv, const float* __restrict__ aggB,
                        float* __restrict__ xc) {
    __shared__ float Bs[256], Cs[256];
    __shared__ int   sperm[16];
    int d = threadIdx.x, chunk = blockIdx.x, base = chunk * 16;
    Bs[d]       = Bb[base * DSTATE + d];
    Bs[128 + d] = Bb[base * DSTATE + 128 + d];
    Cs[d]       = Cb[base * DSTATE + d];
    Cs[128 + d] = Cb[base * DSTATE + 128 + d];
    if (d < 16) sperm[d] = perm_l(dir, base + d);
    float A[DSTATE], h[DSTATE];
#pragma unroll
    for (int n = 0; n < DSTATE; ++n) {
        A[n] = -__expf(A_logs[(dir * 128 + d) * DSTATE + n]);
        h[n] = aggB[(d * DSTATE + n) * NCHUNK + chunk];
    }
    float bias = dtb[dir * 128 + d];
    float Dp   = Dsv[dir * 128 + d];
    __syncthreads();
    for (int t = 0; t < 16; ++t) {
        int l = base + t;
        float u  = xs[l * 128 + d];
        float dt = softplus_f(dtv[l * 128 + d] + bias);
        float du = dt * u;
        float y = 0.f;
#pragma unroll
        for (int n = 0; n < DSTATE; ++n) {
            float a = __expf(dt * A[n]);
            h[n] = a * h[n] + du * Bs[t * 16 + n];
            y += h[n] * Cs[t * 16 + n];
        }
        y += u * Dp;
        xc[sperm[t] * 128 + d] = y;
    }
}

// ---- LayerNorm over channels + SiLU(z) gate, emit f16 for out_proj ----------
__global__ void k_ln(const float* __restrict__ xc, const float* __restrict__ zb,
                     const float* __restrict__ gamma, const float* __restrict__ beta,
                     _Float16* __restrict__ ylh) {
    __shared__ float red[128];
    int l = blockIdx.x, t = threadIdx.x;
    float v = xc[l * 128 + t];
    red[t] = v; __syncthreads();
    for (int s = 64; s > 0; s >>= 1) { if (t < s) red[t] += red[t + s]; __syncthreads(); }
    float mu = red[0] * (1.f / 128.f);
    __syncthreads();
    float dv = v - mu;
    red[t] = dv * dv; __syncthreads();
    for (int s = 64; s > 0; s >>= 1) { if (t < s) red[t] += red[t + s]; __syncthreads(); }
    float var = red[0] * (1.f / 128.f);
    float yn = dv * rsqrtf(var + 1e-5f) * gamma[t] + beta[t];
    ylh[l * 128 + t] = (_Float16)(yn * silu_f(zb[l * 128 + t]));
}

// ---- out_proj GEMM: out[l,c] = sum_e y[l,e]*W[c,e] --------------------------
__global__ void k_outproj(const _Float16* __restrict__ ylh, const float* __restrict__ W,
                          float* __restrict__ out) {
    int tile  = blockIdx.x;
    int wave  = threadIdx.x >> 5;
    int lane  = threadIdx.x & 31;
    int ntile = blockIdx.y * 4 + wave;      // 0..7 (N=128)
    int m = lane & 15, g = lane >> 4;
    const v8h*    Arow = (const v8h*)(ylh + (tile * 16 + m) * 128);
    const float4* Brow = (const float4*)(W + (ntile * 16 + m) * 128);
    v8f acc = {};
#pragma unroll
    for (int kk = 0; kk < 4; ++kk) {
        v8h lo = Arow[kk * 4 + g];           // halves k = kk*32+8g .. +7  (b128)
        v8h hi = Arow[kk * 4 + 2 + g];       // halves k = kk*32+16+8g .. +7
        v16h a = __builtin_shufflevector(lo, hi, 0, 1, 2, 3, 4, 5, 6, 7,
                                         8, 9, 10, 11, 12, 13, 14, 15);
        v16h b = pack_f32(Brow, kk, g);
        acc = __builtin_amdgcn_wmma_f32_16x16x32_f16(false, a, false, b,
                                                     (short)0, acc, false, false);
    }
    float* dst = out + (tile * 16 + 8 * g) * 128 + ntile * 16 + m;
#pragma unroll
    for (int r = 0; r < 8; ++r) dst[r * 128] = acc[r];
}

extern "C" void kernel_launch(void* const* d_in, const int* in_sizes, int n_in,
                              void* d_out, int out_size, void* d_ws, size_t ws_size,
                              hipStream_t stream) {
    const float* x              = (const float*)d_in[0];
    const float* in_proj_w      = (const float*)d_in[1];
    const float* conv_w         = (const float*)d_in[2];
    const float* conv_b         = (const float*)d_in[3];
    const float* x_proj_weight  = (const float*)d_in[4];
    const float* dt_projs_w     = (const float*)d_in[5];
    const float* dt_projs_bias  = (const float*)d_in[6];
    const float* A_logs         = (const float*)d_in[7];
    const float* Ds             = (const float*)d_in[8];
    const float* ln_gamma       = (const float*)d_in[9];
    const float* ln_beta        = (const float*)d_in[10];
    const float* out_proj_w     = (const float*)d_in[11];

    float* ws   = (float*)d_ws;
    float* xp   = ws;                      // [L][128]
    float* zb   = xp   + L_TOT * DI;       // [L][128]
    float* xc   = zb   + L_TOT * DI;       // [L][128] (direction state)
    float* xs   = xc   + L_TOT * DI;       // [L][128] sequence-ordered u
    float* dtv  = xs   + L_TOT * DI;       // [L][128]
    float* Bb   = dtv  + L_TOT * DI;       // [L][16]
    float* Cb   = Bb   + L_TOT * DSTATE;   // [L][16]
    float* aggA = Cb   + L_TOT * DSTATE;   // [2048][864]
    float* aggB = aggA + 2048 * NCHUNK;    // [2048][864]
    _Float16* ylh = (_Float16*)(aggB + 2048 * NCHUNK);  // [L][128] f16

    k_inproj<<<dim3(NCHUNK, 4), 128, 0, stream>>>(x, in_proj_w, xp, zb);
    k_conv<<<L_TOT, 128, 0, stream>>>(xp, conv_w, conv_b, xc);
    for (int dir = 0; dir < 8; ++dir) {
        k_proj<<<NCHUNK, 128, 0, stream>>>(dir, xc, xs,
                                           x_proj_weight + dir * XROWS * DI,
                                           dt_projs_w + dir * DI * DTR,
                                           dtv, Bb, Cb);
        k_scan1<<<NCHUNK, 128, 0, stream>>>(dir, xs, dtv, Bb, A_logs,
                                            dt_projs_bias, aggA, aggB);
        k_chunkscan<<<8, 256, 0, stream>>>(aggA, aggB);
        k_scan3<<<NCHUNK, 128, 0, stream>>>(dir, xs, dtv, Bb, Cb, A_logs,
                                            dt_projs_bias, Ds, aggB, xc);
    }
    k_ln<<<L_TOT, 128, 0, stream>>>(xc, zb, ln_gamma, ln_beta, ylh);
    k_outproj<<<dim3(NCHUNK, 2), 128, 0, stream>>>(ylh, out_proj_w, (float*)d_out);
}
